// MultiHeadAttention_19713899889517
// MI455X (gfx1250) — compile-verified, matrix operand order checked by reading, not run on hardware
//
#include <hip/hip_runtime.h>
#include <hip/hip_bf16.h>
#include <math.h>

#define BB 2
#define SS 2048
#define DD 2048
#define HH 16
#define DKK 128

typedef __bf16 bf16;
typedef __attribute__((ext_vector_type(16))) __bf16 v16bf;
typedef __attribute__((ext_vector_type(8)))  __bf16 v8bf;
typedef __attribute__((ext_vector_type(8)))  float  v8f;

// ---------------------------------------------------------------------------
// Fragment load for 16x32 bf16 A-matrix / 32x16 bf16 B-matrix (CDNA5 WMMA).
// ISA layout (CDNA5 ISA): lane L (L&15 = row for A / col for B), lanes 0-15
// take k = 0..7 & 16..23, lanes 16-31 take k = 8..15 & 24..31, two bf16 per
// VGPR. Source is row-major with leading dim `ld`; both 8-element chunks are
// contiguous 16-byte loads.
// ---------------------------------------------------------------------------
__device__ __forceinline__ v16bf load_frag(const bf16* __restrict__ base, int ld) {
    const int lane = threadIdx.x & 31;
    const int m    = lane & 15;
    const int kb   = (lane & 16) ? 8 : 0;
    const bf16* p  = base + (size_t)m * ld + kb;
    v8bf lo = *reinterpret_cast<const v8bf*>(p);        // k = kb .. kb+7
    v8bf hi = *reinterpret_cast<const v8bf*>(p + 16);   // k = kb+16 .. kb+23
    v16bf r;
#pragma unroll
    for (int i = 0; i < 8; ++i) { r[i] = lo[i]; r[i + 8] = hi[i]; }
    return r;
}

// Per-lane prefetch of the fragment this wave will need PF k-elements later.
__device__ __forceinline__ void prefetch_frag(const bf16* base, int ld) {
    const int lane = threadIdx.x & 31;
    const int m    = lane & 15;
    const int kb   = (lane & 16) ? 8 : 0;
    __builtin_prefetch(base + (size_t)m * ld + kb, 0, 1);  // -> global_prefetch_b8
}

// ---------------------------------------------------------------------------
// f32 -> bf16 bulk convert (vectorized x4)
// ---------------------------------------------------------------------------
__global__ void cvt_bf16(const float* __restrict__ src, bf16* __restrict__ dst, int n) {
    int i = (blockIdx.x * blockDim.x + threadIdx.x) * 4;
    if (i + 3 < n) {
        float4 f = *reinterpret_cast<const float4*>(src + i);
        dst[i + 0] = (bf16)f.x;
        dst[i + 1] = (bf16)f.y;
        dst[i + 2] = (bf16)f.z;
        dst[i + 3] = (bf16)f.w;
    }
}

// ---------------------------------------------------------------------------
// Y = X[M,K] * W[N,K]^T, bf16 in, f32 accumulate via v_wmma_f32_16x16x32_bf16.
// WG = 8 waves, WG tile 128x256, wave tile 64x64 (4x4 C fragments).
// Per k-step: 8 fragment loads (16 x global_load_b128) feed 16 WMMAs.
// mode 0: f32 flat [M,N] -> outF
// mode 1: bf16 [B,H,S,DK] head-split    (Q/K)
// mode 2: bf16 [B,H,DK,S] transposed    (V, so attention B-frags are contig.)
// ---------------------------------------------------------------------------
__global__ __launch_bounds__(256) void gemm_bf16(const bf16* __restrict__ X,
                                                 const bf16* __restrict__ W,
                                                 float* __restrict__ outF,
                                                 bf16* __restrict__ outH,
                                                 int M, int N, int K, int mode) {
    const int wave = threadIdx.x >> 5;
    const int lane = threadIdx.x & 31;
    const int m0   = blockIdx.x * 128 + (wave >> 2) * 64;
    const int n0   = blockIdx.y * 256 + (wave & 3) * 64;

    v8f c[4][4];
#pragma unroll
    for (int i = 0; i < 4; ++i)
#pragma unroll
        for (int j = 0; j < 4; ++j) c[i][j] = (v8f)0.0f;

    const int PF = 128;  // prefetch distance in k-elements (256 B/lane row)
    for (int kk = 0; kk < K; kk += 32) {
        v16bf a[4], b[4];
#pragma unroll
        for (int i = 0; i < 4; ++i)
            a[i] = load_frag(X + (size_t)(m0 + 16 * i) * K + kk, K);
#pragma unroll
        for (int j = 0; j < 4; ++j)
            b[j] = load_frag(W + (size_t)(n0 + 16 * j) * K + kk, K);
        // speculative prefetch of the k-tile PF ahead (OOB is silently dropped)
#pragma unroll
        for (int i = 0; i < 4; ++i)
            prefetch_frag(X + (size_t)(m0 + 16 * i) * K + kk + PF, K);
#pragma unroll
        for (int j = 0; j < 4; ++j)
            prefetch_frag(W + (size_t)(n0 + 16 * j) * K + kk + PF, K);
#pragma unroll
        for (int i = 0; i < 4; ++i)
#pragma unroll
            for (int j = 0; j < 4; ++j)
                c[i][j] = __builtin_amdgcn_wmma_f32_16x16x32_bf16(
                    false, a[i], false, b[j], (short)0, c[i][j], false, false);
    }

    // C layout: VGPR r, lanes 0-15 -> row r, lanes 16-31 -> row r+8; col = lane&15
    const int col = lane & 15;
    const int rb  = (lane & 16) ? 8 : 0;
#pragma unroll
    for (int i = 0; i < 4; ++i)
#pragma unroll
        for (int j = 0; j < 4; ++j)
#pragma unroll
            for (int r = 0; r < 8; ++r) {
                const int gm = m0 + 16 * i + rb + r;
                const int gn = n0 + 16 * j + col;
                const float v = c[i][j][r];
                if (mode == 0) {
                    outF[(size_t)gm * N + gn] = v;
                } else {
                    const int bi = gm >> 11, s = gm & (SS - 1);
                    const int h = gn >> 7, dk = gn & (DKK - 1);
                    if (mode == 1)
                        outH[(((size_t)bi * HH + h) * SS + s) * DKK + dk] = (bf16)v;
                    else
                        outH[(((size_t)bi * HH + h) * DKK + dk) * SS + s] = (bf16)v;
                }
            }
}

// ---------------------------------------------------------------------------
// In-place RoPE on [B,H,S,DK] bf16; one thread per (b,h,s, j<DK/2) pair.
// ---------------------------------------------------------------------------
__global__ void rope_kernel(bf16* __restrict__ t, const int* __restrict__ pos) {
    const int idx = blockIdx.x * blockDim.x + threadIdx.x;  // B*H*S*64 threads
    const int j   = idx & 63;
    const int s   = (idx >> 6) & (SS - 1);
    const int bh  = idx >> 17;
    const float p   = (float)pos[s];
    const float inv = __powf(10000.0f, -(float)(2 * j) / (float)DKK);
    float sn, cs;
    __sincosf(p * inv, &sn, &cs);
    const size_t base = ((size_t)bh * SS + s) * DKK;
    const float x1 = (float)t[base + j];
    const float x2 = (float)t[base + j + 64];
    t[base + j]      = (bf16)(x1 * cs - x2 * sn);
    t[base + j + 64] = (bf16)(x2 * cs + x1 * sn);
}

// ---------------------------------------------------------------------------
// Causal flash attention. One wave (32 threads) per (b, h, 16-row q tile).
// Key blocks of 32: scores = 8 WMMAs, O update = 8 WMMAs via LDS P-relayout.
// q,k: [B,H,S,DK] bf16 (RoPE'd); vt: [B,H,DK,S] bf16; out: [B,S,D] bf16.
// ---------------------------------------------------------------------------
__global__ __launch_bounds__(32) void flash_attn(const bf16* __restrict__ qh,
                                                 const bf16* __restrict__ kh,
                                                 const bf16* __restrict__ vt,
                                                 bf16* __restrict__ ao) {
    __shared__ bf16 pls[16 * 32];  // P tile staging (C-layout -> A-layout)

    const int lane = threadIdx.x & 31;
    const int q0   = blockIdx.x * 16;
    const int h    = blockIdx.y;
    const int b    = blockIdx.z;
    const int bh   = b * HH + h;

    const bf16* qbase = qh + ((size_t)bh * SS + q0) * DKK;
    const bf16* kbase = kh + (size_t)bh * SS * DKK;
    const bf16* vbase = vt + (size_t)bh * DKK * SS;

    v16bf qf[4];
#pragma unroll
    for (int ks = 0; ks < 4; ++ks) qf[ks] = load_frag(qbase + 32 * ks, DKK);

    v8f o[8];
#pragma unroll
    for (int cidx = 0; cidx < 8; ++cidx) o[cidx] = (v8f)0.0f;
    float mrow[8], lrow[8];
#pragma unroll
    for (int r = 0; r < 8; ++r) { mrow[r] = -3.0e38f; lrow[r] = 0.0f; }

    const int col = lane & 15;
    const int rb  = (lane & 16) ? 8 : 0;
    const float scale = 0.08838834764831845f;  // 1/sqrt(128)

    for (int j0 = 0; j0 <= q0 + 15; j0 += 32) {
        // ---- scores: two 16-key tiles, K-dim = 128 in 4 WMMA steps each ----
        v8f sc[2];
#pragma unroll
        for (int t = 0; t < 2; ++t) {
            v8f s = (v8f)0.0f;
#pragma unroll
            for (int ks = 0; ks < 4; ++ks) {
                v16bf kf = load_frag(kbase + (size_t)(j0 + 16 * t) * DKK + 32 * ks, DKK);
                s = __builtin_amdgcn_wmma_f32_16x16x32_bf16(
                    false, qf[ks], false, kf, (short)0, s, false, false);
            }
            const int kidx = j0 + 16 * t + col;
#pragma unroll
            for (int r = 0; r < 8; ++r) {
                const int qrow = q0 + rb + r;
                s[r] = (kidx <= qrow) ? s[r] * scale : -1.0e30f;
            }
            sc[t] = s;
        }

        // ---- online softmax (row reductions across 16-lane half-groups) ----
        float mnew[8], alpha[8];
#pragma unroll
        for (int r = 0; r < 8; ++r) {
            float v = fmaxf(sc[0][r], sc[1][r]);
#pragma unroll
            for (int off = 1; off < 16; off <<= 1)
                v = fmaxf(v, __shfl_xor(v, off, 32));
            mnew[r]  = fmaxf(mrow[r], v);
            alpha[r] = __expf(mrow[r] - mnew[r]);
        }
#pragma unroll
        for (int t = 0; t < 2; ++t)
#pragma unroll
            for (int r = 0; r < 8; ++r) sc[t][r] = __expf(sc[t][r] - mnew[r]);
#pragma unroll
        for (int r = 0; r < 8; ++r) {
            float ps = sc[0][r] + sc[1][r];
#pragma unroll
            for (int off = 1; off < 16; off <<= 1) ps += __shfl_xor(ps, off, 32);
            lrow[r] = lrow[r] * alpha[r] + ps;
            mrow[r] = mnew[r];
        }
#pragma unroll
        for (int cidx = 0; cidx < 8; ++cidx)
#pragma unroll
            for (int r = 0; r < 8; ++r) o[cidx][r] *= alpha[r];

        // ---- P: C-layout -> LDS (16x32 bf16) -> A-fragment ----
#pragma unroll
        for (int t = 0; t < 2; ++t)
#pragma unroll
            for (int r = 0; r < 8; ++r)
                pls[(rb + r) * 32 + t * 16 + col] = (bf16)sc[t][r];
        __syncthreads();

        v16bf pa;
        {
            const int m   = lane & 15;
            const int kb2 = (lane & 16) ? 8 : 0;
            const bf16* pp = pls + m * 32 + kb2;
            v8bf lo = *reinterpret_cast<const v8bf*>(pp);
            v8bf hi = *reinterpret_cast<const v8bf*>(pp + 16);
#pragma unroll
            for (int i = 0; i < 8; ++i) { pa[i] = lo[i]; pa[i + 8] = hi[i]; }
        }

        // ---- O += P(16x32) @ V(32x128): V pre-transposed so B is contiguous
#pragma unroll
        for (int cidx = 0; cidx < 8; ++cidx) {
            v16bf vf = load_frag(vbase + (size_t)(16 * cidx) * SS + j0, SS);
            o[cidx] = __builtin_amdgcn_wmma_f32_16x16x32_bf16(
                false, pa, false, vf, (short)0, o[cidx], false, false);
        }
        __syncthreads();
    }

    // ---- epilogue: normalize and scatter to [B,S,D] bf16 ----
#pragma unroll
    for (int cidx = 0; cidx < 8; ++cidx)
#pragma unroll
        for (int r = 0; r < 8; ++r) {
            const int row  = q0 + rb + r;
            const int dcol = h * DKK + 16 * cidx + col;
            ao[((size_t)b * SS + row) * DD + dcol] = (bf16)(o[cidx][r] / lrow[r]);
        }
}

// ---------------------------------------------------------------------------
extern "C" void kernel_launch(void* const* d_in, const int* in_sizes, int n_in,
                              void* d_out, int out_size, void* d_ws, size_t ws_size,
                              hipStream_t stream) {
    const float* x  = (const float*)d_in[0];
    const float* wq = (const float*)d_in[1];
    const float* wk = (const float*)d_in[2];
    const float* wv = (const float*)d_in[3];
    const float* wo = (const float*)d_in[4];
    const int* pos  = (const int*)d_in[5];
    float* out      = (float*)d_out;

    char* ws = (char*)d_ws;
    size_t off = 0;
    auto walloc = [&](size_t bytes) -> void* {
        void* p = ws + off;
        off += (bytes + 255) & ~(size_t)255;
        return p;
    };
    const size_t xe = (size_t)BB * SS * DD;   // 8.4M elems
    const size_t we = (size_t)DD * DD;        // 4.2M elems
    bf16* xb  = (bf16*)walloc(xe * 2);
    bf16* wqb = (bf16*)walloc(we * 2);
    bf16* wkb = (bf16*)walloc(we * 2);
    bf16* wvb = (bf16*)walloc(we * 2);
    bf16* wob = (bf16*)walloc(we * 2);
    bf16* qhd = (bf16*)walloc(xe * 2);        // [B,H,S,DK]
    bf16* khd = (bf16*)walloc(xe * 2);        // [B,H,S,DK]
    bf16* vtd = (bf16*)walloc(xe * 2);        // [B,H,DK,S]
    bf16* aod = (bf16*)walloc(xe * 2);        // [B,S,D]

    // 1) convert to bf16
    cvt_bf16<<<(int)(xe / 4 / 256), 256, 0, stream>>>(x, xb, (int)xe);
    cvt_bf16<<<(int)(we / 4 / 256), 256, 0, stream>>>(wq, wqb, (int)we);
    cvt_bf16<<<(int)(we / 4 / 256), 256, 0, stream>>>(wk, wkb, (int)we);
    cvt_bf16<<<(int)(we / 4 / 256), 256, 0, stream>>>(wv, wvb, (int)we);
    cvt_bf16<<<(int)(we / 4 / 256), 256, 0, stream>>>(wo, wob, (int)we);

    // 2) QKV projections (WMMA)
    dim3 gg(BB * SS / 128, DD / 256);  // (32, 8)
    gemm_bf16<<<gg, 256, 0, stream>>>(xb, wqb, nullptr, qhd, BB * SS, DD, DD, 1);
    gemm_bf16<<<gg, 256, 0, stream>>>(xb, wkb, nullptr, khd, BB * SS, DD, DD, 1);
    gemm_bf16<<<gg, 256, 0, stream>>>(xb, wvb, nullptr, vtd, BB * SS, DD, DD, 2);

    // 3) RoPE on Q and K
    const int nr = BB * HH * SS * 64;
    rope_kernel<<<nr / 256, 256, 0, stream>>>(qhd, pos);
    rope_kernel<<<nr / 256, 256, 0, stream>>>(khd, pos);

    // 4) causal flash attention (WMMA)
    dim3 ga(SS / 16, HH, BB);
    flash_attn<<<ga, 32, 0, stream>>>(qhd, khd, vtd, aod);

    // 5) output projection (WMMA) -> f32 d_out
    gemm_bf16<<<gg, 256, 0, stream>>>(aod, wob, out, nullptr, BB * SS, DD, DD, 0);
}
